// TernaryConv2d_48524540511058
// MI455X (gfx1250) — matrix operand order repaired and checked by reading, not compiled, not run on hardware
//
#include <hip/hip_runtime.h>
#include <hip/hip_bf16.h>

typedef __attribute__((ext_vector_type(16))) __bf16 v16bf;
typedef __attribute__((ext_vector_type(8)))  __bf16 v8bf;
typedef __attribute__((ext_vector_type(8)))  float  v8f;

// exact parameter type of __builtin_amdgcn_global_load_async_to_lds_b128
// (from the compiler diagnostic: vector_size(16) int, AS1 src / AS3 dst)
typedef int v4i_vs __attribute__((vector_size(16)));
typedef __attribute__((address_space(1))) v4i_vs* async_gptr;
typedef __attribute__((address_space(3))) v4i_vs* async_lptr;

#define WN    589824   // 256*256*3*3 weight elements
#define CIN   256
#define COUT  256
#define HW    56
#define NPIX  3136     // 56*56
#define KDIM  2304     // 256*9
#define NIMG  32
#define RBLK  512      // partial-reduction blocks

#if __has_builtin(__builtin_amdgcn_global_load_async_to_lds_b128) && \
    __has_builtin(__builtin_amdgcn_s_wait_asynccnt)
#define USE_ASYNC_LDS 1
#endif

// ---------------- workspace layout (floats) ----------------
// wsf[0] = sum|w| ; wsf[1] = masked sum ; wsf[2] = masked count ; wsf[3] = alpha
// wsf[8 .. 8+512)        : partials for sum|w|
// wsf[520 .. 520+512)    : partials for masked sum
// wsf[1032 .. 1032+512)  : partials for masked count
// byte offset 8192       : quantized bf16 weights, layout [M=256][r=9][ci=256]

__global__ __launch_bounds__(256) void k_sumabs_part(const float* __restrict__ w,
                                                     float* __restrict__ wsf) {
    __shared__ float red[256];
    float s = 0.f;
    for (int i = blockIdx.x * 256 + threadIdx.x; i < WN; i += RBLK * 256)
        s += fabsf(w[i]);
    red[threadIdx.x] = s;
    __syncthreads();
    for (int off = 128; off > 0; off >>= 1) {
        if (threadIdx.x < off) red[threadIdx.x] += red[threadIdx.x + off];
        __syncthreads();
    }
    if (threadIdx.x == 0) wsf[8 + blockIdx.x] = red[0];
}

__global__ __launch_bounds__(256) void k_sumabs_final(float* __restrict__ wsf) {
    __shared__ float red[256];
    red[threadIdx.x] = wsf[8 + threadIdx.x] + wsf[8 + 256 + threadIdx.x];
    __syncthreads();
    for (int off = 128; off > 0; off >>= 1) {
        if (threadIdx.x < off) red[threadIdx.x] += red[threadIdx.x + off];
        __syncthreads();
    }
    if (threadIdx.x == 0) wsf[0] = red[0];
}

__global__ __launch_bounds__(256) void k_masked_part(const float* __restrict__ w,
                                                     float* __restrict__ wsf) {
    __shared__ float reds[256];
    __shared__ float redc[256];
    const float thr = 0.75f * (wsf[0] / (float)WN);
    float s = 0.f, c = 0.f;
    for (int i = blockIdx.x * 256 + threadIdx.x; i < WN; i += RBLK * 256) {
        float a = fabsf(w[i]);
        if (a > thr) { s += a; c += 1.f; }
    }
    reds[threadIdx.x] = s;
    redc[threadIdx.x] = c;
    __syncthreads();
    for (int off = 128; off > 0; off >>= 1) {
        if (threadIdx.x < off) {
            reds[threadIdx.x] += reds[threadIdx.x + off];
            redc[threadIdx.x] += redc[threadIdx.x + off];
        }
        __syncthreads();
    }
    if (threadIdx.x == 0) {
        wsf[520  + blockIdx.x] = reds[0];
        wsf[1032 + blockIdx.x] = redc[0];
    }
}

__global__ __launch_bounds__(256) void k_masked_final(float* __restrict__ wsf) {
    __shared__ float reds[256];
    __shared__ float redc[256];
    reds[threadIdx.x] = wsf[520  + threadIdx.x] + wsf[520  + 256 + threadIdx.x];
    redc[threadIdx.x] = wsf[1032 + threadIdx.x] + wsf[1032 + 256 + threadIdx.x];
    __syncthreads();
    for (int off = 128; off > 0; off >>= 1) {
        if (threadIdx.x < off) {
            reds[threadIdx.x] += reds[threadIdx.x + off];
            redc[threadIdx.x] += redc[threadIdx.x + off];
        }
        __syncthreads();
    }
    if (threadIdx.x == 0) {
        wsf[1] = reds[0];
        wsf[2] = redc[0];
        wsf[3] = (redc[0] > 0.f) ? (reds[0] / redc[0]) : 1.0f;  // alpha
    }
}

// Quantize weights to exact {-1, 0, +1} bf16, remapped from OIHW k = ci*9 + r
// into GEMM-friendly k' = r*256 + ci  (layout [M][9][256]).
__global__ __launch_bounds__(256) void k_quant(const float* __restrict__ w,
                                               const float* __restrict__ wsf,
                                               __bf16* __restrict__ wq) {
    const float thr = 0.75f * (wsf[0] / (float)WN);
    int i = blockIdx.x * 256 + threadIdx.x;
    if (i < WN) {
        int m   = i / KDIM;
        int rem = i - m * KDIM;
        int ci  = rem / 9;
        int r   = rem - ci * 9;
        float v = w[i];
        float t = (v > thr) ? 1.f : ((v < -thr) ? -1.f : 0.f);
        wq[(size_t)m * KDIM + r * 256 + ci] = (__bf16)t;
    }
}

// ---------------- implicit-GEMM conv: bf16 WMMA ----------------
// Block tile: 256(M = all out-ch) x 64(N = pixels). K ordered as 9 x 256(ci),
// stepped 32 at a time. 8 waves: wave w owns M rows [w*32, w*32+32) x all 64 N
// -> 2(m) x 4(n) accumulators, 8 wmma per K-step.
__global__ __launch_bounds__(256) void k_conv(const float* __restrict__ x,
                                              const __bf16* __restrict__ wq,
                                              const float* __restrict__ wsf,
                                              float* __restrict__ out) {
    __shared__ __bf16 As[256][32];  // [m][k_local]
    __shared__ __bf16 Bs[64][32];   // [n][k_local]

    const int tid  = threadIdx.x;
    const int lane = tid & 31;
    const int wave = tid >> 5;

    const int pixBase = blockIdx.x * 64;
    const int img     = blockIdx.y;
    const float* __restrict__ xImg = x + (size_t)img * CIN * NPIX;

    // B-staging thread-invariants: thread covers (nl, kl0 + 4*it), it = 0..7
    const int bnl  = tid & 63;       // pixel within tile (fast -> coalesced x reads)
    const int bkl0 = tid >> 6;       // 0..3
    const int bpix = pixBase + bnl;
    const int bh   = bpix / HW;
    const int bw   = bpix - bh * HW;

    v8f acc[2][4];
    const v8f vzero = {0.f, 0.f, 0.f, 0.f, 0.f, 0.f, 0.f, 0.f};
    #pragma unroll
    for (int mt = 0; mt < 2; mt++)
        #pragma unroll
        for (int nt = 0; nt < 4; nt++) acc[mt][nt] = vzero;

    for (int r = 0; r < 9; ++r) {            // (kh, kw) taps
        const int kh = r / 3;
        const int kw = r - kh * 3;
        const int ih = bh + kh - 1;
        const int iw = bw + kw - 1;
        const bool valid = ((unsigned)ih < (unsigned)HW) && ((unsigned)iw < (unsigned)HW);
        const float* __restrict__ bsrc = xImg + (ih * HW + iw);  // base for ci=0

        for (int cb = 0; cb < 8; ++cb) {     // ci blocks of 32
            const int ks = r * 256 + cb * 32;

            // stage A tile: 256x32 bf16, 16B chunks (4 per thread).
            // CDNA5 async global->LDS path: no VGPR round-trip, copy engine
            // overlaps with the B gather below. Fallback: manual copy.
            #pragma unroll
            for (int j = 0; j < 4; ++j) {
                int t   = tid + j * 256;
                int row = t >> 2;
                int ch  = t & 3;
#ifdef USE_ASYNC_LDS
                __builtin_amdgcn_global_load_async_to_lds_b128(
                    (async_gptr)(wq + (size_t)row * KDIM + ks + ch * 8),
                    (async_lptr)&As[row][ch * 8],
                    0, 0);
#else
                *(v8bf*)&As[row][ch * 8] =
                    *(const v8bf*)(wq + (size_t)row * KDIM + ks + ch * 8);
#endif
            }
            // stage B tile: shifted-x gather (f32 -> bf16), index math hoisted
            #pragma unroll
            for (int it = 0; it < 8; ++it) {
                int kl = bkl0 + it * 4;
                float v = 0.f;
                if (valid) v = bsrc[(size_t)(cb * 32 + kl) * NPIX];
                Bs[bnl][kl] = (__bf16)v;
            }
#ifdef USE_ASYNC_LDS
            __builtin_amdgcn_s_wait_asynccnt(0);
#endif
            __syncthreads();

            // A fragment (16x32 bf16): lane<16 -> row=lane,    K {0..7,16..23}
            //                          lane>=16 -> row=lane-16, K {8..15,24..31}
            v16bf a[2], b[4];
            const int arow = wave * 32 + (lane & 15);
            const int aklo = (lane >> 4) ? 8 : 0;
            #pragma unroll
            for (int mt = 0; mt < 2; mt++) {
                v8bf lo = *(const v8bf*)&As[arow + mt * 16][aklo];
                v8bf hi = *(const v8bf*)&As[arow + mt * 16][aklo + 16];
                #pragma unroll
                for (int e = 0; e < 8; e++) { a[mt][e] = lo[e]; a[mt][e + 8] = hi[e]; }
            }
            // B fragment (32x16): col = lane%16, K {0..7,16..23} / {8..15,24..31} by lane/16
            const int bcol = lane & 15;
            const int bklo = (lane >> 4) * 8;
            #pragma unroll
            for (int nt = 0; nt < 4; nt++) {
                v8bf lo = *(const v8bf*)&Bs[nt * 16 + bcol][bklo];
                v8bf hi = *(const v8bf*)&Bs[nt * 16 + bcol][bklo + 16];
                #pragma unroll
                for (int e = 0; e < 8; e++) { b[nt][e] = lo[e]; b[nt][e + 8] = hi[e]; }
            }

            #pragma unroll
            for (int mt = 0; mt < 2; mt++)
                #pragma unroll
                for (int nt = 0; nt < 4; nt++)
                    acc[mt][nt] = __builtin_amdgcn_wmma_f32_16x16x32_bf16(
                        false, a[mt], false, b[nt], (short)0, acc[mt][nt], false, false);

            __syncthreads();
        }
    }

    // epilogue: scale by alpha; C/D layout: N = lane%16, M = 8*(lane/16) + v
    const float alpha = wsf[3];
    const int crow0 = (lane >> 4) * 8;
    const int ccol  = lane & 15;
    #pragma unroll
    for (int mt = 0; mt < 2; mt++) {
        #pragma unroll
        for (int nt = 0; nt < 4; nt++) {
            int pix = pixBase + nt * 16 + ccol;
            int m0  = wave * 32 + mt * 16 + crow0;
            float* op = out + ((size_t)img * COUT + m0) * NPIX + pix;
            #pragma unroll
            for (int v = 0; v < 8; v++)
                op[(size_t)v * NPIX] = acc[mt][nt][v] * alpha;
        }
    }
}

extern "C" void kernel_launch(void* const* d_in, const int* in_sizes, int n_in,
                              void* d_out, int out_size, void* d_ws, size_t ws_size,
                              hipStream_t stream) {
    const float* x = (const float*)d_in[0];
    const float* w = (const float*)d_in[1];
    float* out = (float*)d_out;
    float* wsf = (float*)d_ws;
    __bf16* wq = (__bf16*)((char*)d_ws + 8192);

    k_sumabs_part <<<RBLK, 256, 0, stream>>>(w, wsf);
    k_sumabs_final<<<1,    256, 0, stream>>>(wsf);
    k_masked_part <<<RBLK, 256, 0, stream>>>(w, wsf);
    k_masked_final<<<1,    256, 0, stream>>>(wsf);
    k_quant       <<<(WN + 255) / 256, 256, 0, stream>>>(w, wsf, wq);

    dim3 grid(NPIX / 64, NIMG);  // 49 x 32
    k_conv<<<grid, 256, 0, stream>>>(x, wq, wsf, out);
}
